// My_anchor_Loss_30855045054677
// MI455X (gfx1250) — matrix operand order repaired and checked by reading, not compiled, not run on hardware
//
#include <hip/hip_runtime.h>

#define NV   2
#define NB   16384
#define NC   100
#define ND   512
#define CPAD 112   // 7 tiles of 16 rows

typedef _Float16 h8  __attribute__((ext_vector_type(8)));
typedef _Float16 v16h __attribute__((ext_vector_type(16)));
typedef float    v8f  __attribute__((ext_vector_type(8)));

#define MARGIN 10.0f
#define EPSV   1e-8f

__device__ __forceinline__ float wave_sum32(float x) {
    #pragma unroll
    for (int m = 16; m > 0; m >>= 1) x += __shfl_xor(x, m, 32);
    return x;
}

// ---------------------------------------------------------------------------
// Kernel 1: convert anchor f32 -> f16 padded to CPAD rows; zero accumulators.
// grid: V*CPAD*ND / 256 = 448 blocks
// ---------------------------------------------------------------------------
__global__ void prep_kernel(const float* __restrict__ anchor,
                            _Float16* __restrict__ anc_h,
                            float* __restrict__ accums /* 4 floats */) {
    int idx = blockIdx.x * blockDim.x + threadIdx.x;
    if (idx < 4) accums[idx] = 0.0f;
    int v = idx / (CPAD * ND);
    int rem = idx - v * (CPAD * ND);
    int c = rem / ND;
    int d = rem - c * ND;
    float val = (c < NC) ? anchor[((size_t)v * NC + c) * ND + d] : 0.0f;
    anc_h[idx] = (_Float16)val;
}

// ---------------------------------------------------------------------------
// Kernel 2: sq[v,c] = sum_d anchor^2 (f32). One wave per (v,c): 200 waves.
// grid: 25 blocks x 256
// ---------------------------------------------------------------------------
__global__ void sq_kernel(const float* __restrict__ anchor,
                          float* __restrict__ sq) {
    int w = blockIdx.x * 8 + (threadIdx.x >> 5);   // 0..199
    int lane = threadIdx.x & 31;
    const float* row = anchor + (size_t)w * ND;
    float acc = 0.0f;
    #pragma unroll
    for (int j = 0; j < 4; ++j) {
        float4 a = *(const float4*)(row + j * 128 + lane * 4);
        acc += a.x * a.x + a.y * a.y + a.z * a.z + a.w * a.w;
    }
    acc = wave_sum32(acc);
    if (lane == 0) sq[w] = acc;
}

// ---------------------------------------------------------------------------
// Kernel 3: main bandwidth-bound pass. One wave per (v,b): 32768 waves.
// grid: 4096 blocks x 256.  sims -> d_out[1 + v*NB + b]; center-loss atomics.
// ---------------------------------------------------------------------------
__global__ void main_kernel(const float* __restrict__ feat,
                            const int*   __restrict__ label,
                            const float* __restrict__ anchor,
                            const float* __restrict__ sq,
                            float* __restrict__ out,
                            float* __restrict__ acc_cl /* 2 floats */) {
    int w = blockIdx.x * 8 + (threadIdx.x >> 5);   // 0..32767
    int lane = threadIdx.x & 31;
    int v = w >> 14;            // / NB
    int b = w & (NB - 1);
    int c = label[b];

    const float* frow = feat   + (size_t)w * ND;
    const float* arow = anchor + ((size_t)v * NC + c) * ND;

    float ff = 0.0f, fc = 0.0f;
    #pragma unroll
    for (int j = 0; j < 4; ++j) {
        float4 f = *(const float4*)(frow + j * 128 + lane * 4);
        float4 a = *(const float4*)(arow + j * 128 + lane * 4);
        ff += f.x * f.x + f.y * f.y + f.z * f.z + f.w * f.w;
        fc += f.x * a.x + f.y * a.y + f.z * a.z + f.w * a.w;
    }
    ff = wave_sum32(ff);
    fc = wave_sum32(fc);

    if (lane == 0) {
        float cc = sq[v * NC + c];
        float fn = fmaxf(sqrtf(ff), EPSV);
        float cn = fmaxf(sqrtf(cc), EPSV);
        out[1 + w] = fc / (fn * cn);
        atomicAdd(&acc_cl[v], ff - 2.0f * fc + cc);
    }
}

// ---------------------------------------------------------------------------
// Kernel 4: Gram via WMMA f16->f32. One wave per (v, mtile, ntile): 98 waves.
// grid: 98 blocks x 32.  Accumulates sum of off-diagonal gram into acc_off[v].
// ---------------------------------------------------------------------------
__global__ void gram_kernel(const _Float16* __restrict__ anc_h,
                            float* __restrict__ acc_off /* 2 floats */) {
    int t = blockIdx.x;             // 0..97
    int v = t / 49;
    int rem = t - v * 49;
    int mt = rem / 7;
    int nt = rem - mt * 7;
    int c0 = mt * 16, e0 = nt * 16;

    int lane = threadIdx.x;         // 0..31
    int lrow = lane & 15;
    int hsel = lane >> 4;           // 0 | 1

    const _Float16* base = anc_h + (size_t)v * CPAD * ND;
    const _Float16* arow = base + (size_t)(c0 + lrow) * ND;  // A rows (M)
    const _Float16* brow = base + (size_t)(e0 + lrow) * ND;  // B cols (N)

    v8f acc = {};
    #pragma unroll 4
    for (int k0 = 0; k0 < ND; k0 += 32) {
        // A fragment 16x32: lane<16 -> K 0..7 & 16..23 ; lane>=16 -> K 8..15 & 24..31
        h8 alo = *(const h8*)(arow + k0 + hsel * 8);
        h8 ahi = *(const h8*)(arow + k0 + 16 + hsel * 8);
        v16h a = __builtin_shufflevector(alo, ahi,
                    0, 1, 2, 3, 4, 5, 6, 7, 8, 9, 10, 11, 12, 13, 14, 15);
        // B fragment 32x16: B[k,n] = anc[e0+n, d0+k]; lane holds N=lane%16,
        // K 0..7 & 16..23 (lane<16) or 8..15 & 24..31 (lane>=16)
        h8 blo = *(const h8*)(brow + k0 + hsel * 8);
        h8 bhi = *(const h8*)(brow + k0 + 16 + hsel * 8);
        v16h bfrag = __builtin_shufflevector(blo, bhi,
                    0, 1, 2, 3, 4, 5, 6, 7, 8, 9, 10, 11, 12, 13, 14, 15);
        acc = __builtin_amdgcn_wmma_f32_16x16x32_f16(
                  false, a, false, bfrag, (short)0, acc, false, false);
    }

    // C/D layout: lane l holds N = l%16, rows M = r (+8 for lanes 16..31)
    float off = 0.0f;
    int e = e0 + (lane & 15);
    #pragma unroll
    for (int r = 0; r < 8; ++r) {
        int c = c0 + r + ((lane < 16) ? 0 : 8);
        float val = acc[r];
        if (c < NC && e < NC && c != e) off += val;
    }
    off = wave_sum32(off);
    if (lane == 0) atomicAdd(&acc_off[v], off);
}

// ---------------------------------------------------------------------------
// Kernel 5: finish — combine everything into the scalar loss.
// ---------------------------------------------------------------------------
__global__ void finish_kernel(const float* __restrict__ sq,
                              const float* __restrict__ acc_cl,
                              const float* __restrict__ acc_off,
                              float* __restrict__ out) {
    float loss = 0.0f;
    for (int v = 0; v < NV; ++v) {
        float sqsum = 0.0f;
        for (int c = 0; c < NC; ++c) sqsum += sq[v * NC + c];
        float cl = acc_cl[v] / (2.0f * (float)NB);
        // pair_sum = (C-1)*sum(sq) - sum_{c!=e} gram
        float pair_sum = (float)(NC - 1) * sqsum - acc_off[v];
        float inter = pair_sum / (float)(NC * (NC - 1));
        float il = fmaxf(MARGIN - inter, 0.0f);
        loss += 0.5f * cl + 0.5f * il;
    }
    out[0] = loss;
}

// ---------------------------------------------------------------------------
extern "C" void kernel_launch(void* const* d_in, const int* in_sizes, int n_in,
                              void* d_out, int out_size, void* d_ws, size_t ws_size,
                              hipStream_t stream) {
    const float* feat   = (const float*)d_in[0];   // (2,16384,512) f32
    const int*   label  = (const int*)d_in[1];     // (16384,) int
    const float* anchor = (const float*)d_in[2];   // (2,100,512) f32
    float* out = (float*)d_out;                    // [loss, sims(2,16384)]

    // workspace layout (floats)
    float* wsf      = (float*)d_ws;
    float* sq       = wsf;          // 200
    float* acc      = wsf + 200;    // acc_cl[2], acc_off[2]
    float* acc_cl   = acc;
    float* acc_off  = acc + 2;
    _Float16* anc_h = (_Float16*)(wsf + 256);  // V*CPAD*ND halves (~224 KB)

    // 1) f32->f16 padded anchor + zero accumulators
    prep_kernel<<<(NV * CPAD * ND) / 256, 256, 0, stream>>>(anchor, anc_h, acc);
    // 2) per-anchor-row squared norms (f32)
    sq_kernel<<<(NV * NC) / 8, 256, 0, stream>>>(anchor, sq);
    // 3) bandwidth-bound sims + center-loss pass
    main_kernel<<<(NV * NB) / 8, 256, 0, stream>>>(feat, label, anchor, sq,
                                                   out, acc_cl);
    // 4) Gram matrix via v_wmma_f32_16x16x32_f16
    gram_kernel<<<NV * 7 * 7, 32, 0, stream>>>(anc_h, acc_off);
    // 5) scalar loss
    finish_kernel<<<1, 1, 0, stream>>>(sq, acc_cl, acc_off, out);
}
